// LinearAttention_8065948582522
// MI455X (gfx1250) — matrix operand order repaired and checked by reading, not compiled, not run on hardware
//
#include <hip/hip_runtime.h>

// MI455X / gfx1250: wave32, WMMA 16x16x32 f16 -> f32 accumulate.
// Pipeline is HBM-bound (~100MB @ 23.3TB/s); all GEMMs use v_wmma_f32_16x16x32_f16.
// All WMMA operands are fetched K-major so every gather is 2x b128 loads.

typedef _Float16 half_t;
typedef __attribute__((ext_vector_type(16))) _Float16 v16h;
typedef __attribute__((ext_vector_type(8)))  float    v8f;

#define BATCH 16
#define NPIX  4096           // 64*64
#define C_IN  128
#define F_QKV 384
#define M_TOT (BATCH * NPIX) // 65536

__device__ inline v8f wmma_f16(v16h a, v16h b, v8f c) {
  return __builtin_amdgcn_wmma_f32_16x16x32_f16(false, a, false, b, (short)0, c,
                                                false, false);
}

// 16 contiguous halves -> v16h (2x b128). Works for global and LDS pointers.
__device__ inline v16h load16h(const half_t* p) {
  union { uint4 q[2]; v16h h; } u;
  u.q[0] = *(const uint4*)p;
  u.q[1] = *(const uint4*)(p + 8);
  return u.h;
}

// A-operand: 16 contiguous halves at p (K{kb..kb+15}, same order as B)
__device__ inline v16h load16h_a(const half_t* p) {
  union { uint4 q[2]; v16h h; } u;
  u.q[0] = *(const uint4*)p;
  u.q[1] = *(const uint4*)(p + 8);
  return u.h;
}

// ------------------------------------------------- P0: weights -> f16, TRANSPOSED [N][K]
__global__ void cvt_weights(const float* __restrict__ wq, const float* __restrict__ wo,
                            half_t* __restrict__ wqt, half_t* __restrict__ wot) {
  int i = blockIdx.x * 256 + threadIdx.x;
  if (i < C_IN * F_QKV) {
    int k = i / F_QKV, n = i % F_QKV;
    wqt[(size_t)n * C_IN + k] = (half_t)wq[i];
  }
  if (i < C_IN * C_IN) {
    int k = i / C_IN, n = i % C_IN;
    wot[(size_t)n * C_IN + k] = (half_t)wo[i];
  }
}

// ------------------------------------------------- P1: qkv = x @ w_qkv
// block 256 thr (8 waves), tile M=64 x N=96, K=128 in 4 chunks of 32.
// A staged once in LDS (f32->f16, reused by 6 N-tiles); B direct from L2-hot wq_t.
// q row-major f16 [M,128]; k,v stored TRANSPOSED f16 [b][c][n] (packed b128 stores).
__global__ __launch_bounds__(256) void qkv_gemm(const float* __restrict__ x,
                                                const half_t* __restrict__ wqt,
                                                half_t* __restrict__ qh,
                                                half_t* __restrict__ kth,
                                                half_t* __restrict__ vth) {
  __shared__ __align__(16) half_t Ash[64][32];
  const int tid = threadIdx.x;
  const int lane = tid & 31;
  const int wv = tid >> 5;
  const int m0 = blockIdx.x * 64;
  const int n0 = blockIdx.y * 96;
  const int mi = wv & 3;            // M sub-tile 0..3
  const int nset = (wv >> 2) * 3;   // N sub-tiles nset..nset+2

  v8f zero = {};
  v8f acc[3] = {zero, zero, zero};

  const int am = lane & 15, hi = lane >> 4;
  const int ko = hi * 8, kb = hi * 16, bn = lane & 15;

  for (int kc = 0; kc < 4; ++kc) {
    { // stage A: 64x32 f32 -> f16 LDS, one b128 store per thread
      int row = tid >> 2;
      int col = (tid & 3) * 8;
      const float* src = x + (size_t)(m0 + row) * C_IN + kc * 32 + col;
      __builtin_prefetch(src + 32, 0, 0);
      float4 f0 = *(const float4*)src;
      float4 f1 = *(const float4*)(src + 4);
      union { uint4 u; half_t h[8]; } p;
      p.h[0] = (half_t)f0.x; p.h[1] = (half_t)f0.y; p.h[2] = (half_t)f0.z; p.h[3] = (half_t)f0.w;
      p.h[4] = (half_t)f1.x; p.h[5] = (half_t)f1.y; p.h[6] = (half_t)f1.z; p.h[7] = (half_t)f1.w;
      *(uint4*)&Ash[row][col] = p.u;
    }
    __syncthreads();

    v16h a = load16h_a(&Ash[mi * 16 + am][kb]);
    for (int j = 0; j < 3; ++j) {
      int ni = nset + j;
      v16h b = load16h(wqt + (size_t)(n0 + ni * 16 + bn) * C_IN + kc * 32 + kb);
      acc[j] = wmma_f16(a, b, acc[j]);
    }
    __syncthreads();
  }

  // epilogue: D tile lane holds col (lane&15), rows r+8*hi (consecutive pixels)
  for (int j = 0; j < 3; ++j) {
    int ni = nset + j;
    int c = n0 + ni * 16 + bn;
    int mrow0 = m0 + mi * 16 + 8 * hi;
    if (c < 128) {
      for (int r = 0; r < 8; ++r)
        qh[(size_t)(mrow0 + r) * C_IN + c] = (half_t)acc[j][r];
    } else {
      int cc = c & 127;
      half_t* dst = (c < 256) ? kth : vth;
      int bb = mrow0 >> 12;
      int nn = mrow0 & 4095;                 // 8 consecutive n -> one b128 store
      union { uint4 u; half_t h[8]; } p;
      for (int r = 0; r < 8; ++r) p.h[r] = (half_t)acc[j][r];
      *(uint4*)&dst[((size_t)bb * C_IN + cc) * NPIX + nn] = p.u;
    }
  }
}

// ------------------------------------------------- P2: k column stats
// one block per (b,c): max + sumexp over n=4096 (contiguous in k^T)
__global__ __launch_bounds__(256) void kstats(const half_t* __restrict__ kth,
                                              float* __restrict__ stats) {
  __shared__ float red[256];
  int g = blockIdx.x;
  const half_t* p = kth + (size_t)g * NPIX;
  int t = threadIdx.x;
  float mx = -1e30f;
  for (int i = t; i < NPIX; i += 256) mx = fmaxf(mx, (float)p[i]);
  red[t] = mx; __syncthreads();
  for (int s = 128; s > 0; s >>= 1) {
    if (t < s) red[t] = fmaxf(red[t], red[t + s]);
    __syncthreads();
  }
  mx = red[0]; __syncthreads();
  float sm = 0.f;
  for (int i = t; i < NPIX; i += 256) sm += __expf((float)p[i] - mx);
  red[t] = sm; __syncthreads();
  for (int s = 128; s > 0; s >>= 1) {
    if (t < s) red[t] += red[t + s];
    __syncthreads();
  }
  if (t == 0) { stats[g * 2] = mx; stats[g * 2 + 1] = red[0]; }
}

// ------------------------------------------------- P3: context GEMM (no LDS, no barriers)
// per (b,h): softk[32x4096] @ v^T[4096x32]; 4 waves = 4 independent 16x16 tiles.
// A rows contiguous in k^T (exp fused); B columns contiguous in v^T.
// Output written TRANSPOSED f16 ctx_t[bh][e][d] with packed b128 stores.
__global__ __launch_bounds__(128) void ctx_gemm(const half_t* __restrict__ kth,
                                                const half_t* __restrict__ vth,
                                                const float* __restrict__ stats,
                                                half_t* __restrict__ ctxt) {
  int bh = blockIdx.x;           // 0..63
  int b = bh >> 2, h = bh & 3;
  int tid = threadIdx.x, lane = tid & 31, wv = tid >> 5;
  int mi = wv >> 1, ni = wv & 1;
  int am = lane & 15, hi = lane >> 4, ko = hi * 8, kb = hi * 16, bn = lane & 15;
  int d = mi * 16 + am;          // A row within head
  const half_t* arow = kth + ((size_t)b * C_IN + h * 32 + d) * NPIX;
  const half_t* brow = vth + ((size_t)b * C_IN + h * 32 + ni * 16 + bn) * NPIX;
  float rmax = stats[((size_t)b * C_IN + h * 32 + d) * 2];
  v8f acc = {};
  for (int kc = 0; kc < 128; ++kc) {
    int nb = kc * 32;
    __builtin_prefetch(arow + nb + 64, 0, 0);
    __builtin_prefetch(brow + nb + 64, 0, 0);
    union { uint4 u; half_t h[8]; } p0, p1;
    p0.u = *(const uint4*)(arow + nb + kb);
    p1.u = *(const uint4*)(arow + nb + kb + 8);
    v16h a;
    for (int i = 0; i < 8; ++i) {           // fused exp(k - max)
      a[i]     = (half_t)__expf((float)p0.h[i] - rmax);
      a[i + 8] = (half_t)__expf((float)p1.h[i] - rmax);
    }
    v16h bm = load16h(brow + nb + kb);
    acc = wmma_f16(a, bm, acc);
  }
  // fold 1/sum (row scaling), store transposed [e][d]: 8 consecutive d per lane
  union { uint4 u; half_t h[8]; } p;
  for (int r = 0; r < 8; ++r) {
    int dd = mi * 16 + 8 * hi + r;
    float inv = 1.0f / stats[((size_t)b * C_IN + h * 32 + dd) * 2 + 1];
    p.h[r] = (half_t)(acc[r] * inv);
  }
  *(uint4*)&ctxt[((size_t)bh * 32 + ni * 16 + bn) * 32 + mi * 16 + 8 * hi] = p.u;
}

// ------------------------------------------------- P4: fused tail
// per 16-pixel strip: q-softmax -> attn apply (WMMA) -> out proj (WMMA) -> bias+LN
__global__ __launch_bounds__(128) void attn_out(const half_t* __restrict__ qh,
                                                const half_t* __restrict__ ctxt,
                                                const half_t* __restrict__ wot,
                                                const float* __restrict__ bout,
                                                const float* __restrict__ gln,
                                                const float* __restrict__ bln,
                                                float* __restrict__ out) {
  __shared__ __align__(16) half_t qs[16][128];
  __shared__ __align__(16) half_t att[16][128];
  __shared__ float outf[16][128];
  __shared__ float wsum[16][8], wsq[16][8];
  __shared__ float mu[16], rs[16];
  int tid = threadIdx.x, lane = tid & 31, wv = tid >> 5;
  int gm0 = blockIdx.x * 16;
  int b = gm0 >> 12;
  int am = lane & 15, hi = lane >> 4, ko = hi * 8, kb = hi * 16, bn = lane & 15;

  // q softmax over d=32 per (row, head); groups are contiguous in q
  if (tid < 64) {
    int row = tid >> 2, h = tid & 3;
    const half_t* src = qh + (size_t)(gm0 + row) * C_IN + h * 32;
    float v[32]; float mx = -1e30f;
    for (int i = 0; i < 32; ++i) { v[i] = (float)src[i]; mx = fmaxf(mx, v[i]); }
    float s = 0.f;
    for (int i = 0; i < 32; ++i) { v[i] = __expf(v[i] - mx); s += v[i]; }
    float inv = 0.17677669529663689f / s;   // scale = 32^-0.5 folded in
    union { uint4 u; half_t h[8]; } p;
    for (int c = 0; c < 4; ++c) {
      for (int i = 0; i < 8; ++i) p.h[i] = (half_t)(v[c * 8 + i] * inv);
      *(uint4*)&qs[row][h * 32 + c * 8] = p.u;
    }
  }
  __syncthreads();

  { // attn apply: head h = wave; B from transposed ctx (K=d contiguous)
    int h = wv;
    v16h a = load16h_a(&qs[am][h * 32 + kb]);
    const half_t* cb = ctxt + (size_t)(b * 4 + h) * 1024;   // [e][d] 32x32
    v16h b0 = load16h(cb + (size_t)bn * 32 + kb);
    v16h b1 = load16h(cb + (size_t)(16 + bn) * 32 + kb);
    v8f c0 = {}, c1 = {};
    c0 = wmma_f16(a, b0, c0);
    c1 = wmma_f16(a, b1, c1);
    for (int r = 0; r < 8; ++r) {
      att[r + 8 * hi][h * 32 + bn]      = (half_t)c0[r];
      att[r + 8 * hi][h * 32 + 16 + bn] = (half_t)c1[r];
    }
  }
  __syncthreads();

  { // out projection: [16x128] @ w_out; B direct from L2-hot transposed weights
    for (int jt = 0; jt < 2; ++jt) {
      int nt = wv * 2 + jt;
      v8f acc = {};
      for (int kc = 0; kc < 4; ++kc) {
        v16h a = load16h_a(&att[am][kc * 32 + kb]);
        v16h bm = load16h(wot + (size_t)(nt * 16 + bn) * C_IN + kc * 32 + kb);
        acc = wmma_f16(a, bm, acc);
      }
      int col = nt * 16 + bn;
      float bo = bout[col];
      for (int r = 0; r < 8; ++r) outf[r + 8 * hi][col] = acc[r] + bo;
    }
  }
  __syncthreads();

  { // LayerNorm over 128 channels per row
    int row = tid >> 3, j = tid & 7;
    float s = 0.f, q2 = 0.f;
    for (int c = j * 16; c < j * 16 + 16; ++c) {
      float v = outf[row][c]; s += v; q2 += v * v;
    }
    wsum[row][j] = s; wsq[row][j] = q2;
  }
  __syncthreads();
  if (tid < 16) {
    float s = 0.f, q2 = 0.f;
    for (int j = 0; j < 8; ++j) { s += wsum[tid][j]; q2 += wsq[tid][j]; }
    float m = s * (1.f / 128.f);
    float var = q2 * (1.f / 128.f) - m * m;
    mu[tid] = m; rs[tid] = rsqrtf(var + 1e-5f);
  }
  __syncthreads();
  {
    int row = tid >> 3, j = tid & 7;
    float m = mu[row], r = rs[row];
    float* dst = out + (size_t)(gm0 + row) * C_IN;
    for (int c = j * 16; c < j * 16 + 16; ++c)
      dst[c] = (outf[row][c] - m) * r * gln[c] + bln[c];
  }
}

// ------------------------------------------------- host
extern "C" void kernel_launch(void* const* d_in, const int* in_sizes, int n_in,
                              void* d_out, int out_size, void* d_ws, size_t ws_size,
                              hipStream_t stream) {
  (void)in_sizes; (void)n_in; (void)out_size; (void)ws_size;
  const float* x     = (const float*)d_in[0];
  const float* w_qkv = (const float*)d_in[1];
  const float* w_out = (const float*)d_in[2];
  const float* b_out = (const float*)d_in[3];
  const float* g_ln  = (const float*)d_in[4];
  const float* b_ln  = (const float*)d_in[5];
  float* out = (float*)d_out;

  char* ws = (char*)d_ws;
  size_t off = 0;
  auto carve = [&](size_t bytes) -> void* {
    void* p = ws + off;
    off = (off + bytes + 255) & ~(size_t)255;
    return p;
  };
  half_t* wqt   = (half_t*)carve((size_t)C_IN * F_QKV * 2);   // [384][128]
  half_t* wot   = (half_t*)carve((size_t)C_IN * C_IN * 2);    // [128][128]
  half_t* qh    = (half_t*)carve((size_t)M_TOT * C_IN * 2);   // [M][128]
  half_t* kth   = (half_t*)carve((size_t)M_TOT * C_IN * 2);   // [b][c][n]
  half_t* vth   = (half_t*)carve((size_t)M_TOT * C_IN * 2);   // [b][c][n]
  float*  stats = (float*)carve((size_t)BATCH * C_IN * 2 * 4);
  half_t* ctxt  = (half_t*)carve((size_t)BATCH * 4 * 32 * 32 * 2); // [bh][e][d]

  cvt_weights<<<dim3((C_IN * F_QKV + 255) / 256), dim3(256), 0, stream>>>(
      w_qkv, w_out, wqt, wot);
  qkv_gemm<<<dim3(M_TOT / 64, F_QKV / 96), dim3(256), 0, stream>>>(
      x, wqt, qh, kth, vth);
  kstats<<<dim3(BATCH * C_IN), dim3(256), 0, stream>>>(kth, stats);
  ctx_gemm<<<dim3(BATCH * 4), dim3(128), 0, stream>>>(kth, vth, stats, ctxt);
  attn_out<<<dim3(M_TOT / 16), dim3(128), 0, stream>>>(
      qh, ctxt, wot, b_out, g_ln, b_ln, out);
}
